// CrossAttention_76390288327205
// MI455X (gfx1250) — compile-verified, matrix operand order checked by reading, not run on hardware
//
#include <hip/hip_runtime.h>

#define BB   2
#define SS   2048
#define KVN  2048
#define EE   1024
#define HH   16
#define DD   64
#define NBK  32

typedef __attribute__((ext_vector_type(16))) __bf16 v16bf;
typedef __attribute__((ext_vector_type(8)))  __bf16 v8bf;
typedef __attribute__((ext_vector_type(8)))  float  v8f;

union Frag { v16bf v; v8bf h8[2]; };

static __device__ __forceinline__ v8f wmma_bf16(Frag a, Frag b, v8f c) {
  return __builtin_amdgcn_wmma_f32_16x16x32_bf16(false, a.v, false, b.v,
                                                 (short)0, c, false, false);
}

// ---------------- conversion kernels ----------------

__global__ void k_cvt_bf16(const float* __restrict__ in, __bf16* __restrict__ out, int n) {
  int i = blockIdx.x * blockDim.x + threadIdx.x;
  if (i < n) out[i] = (__bf16)in[i];
}

// in: fp32 [Kd, Nd] row-major  ->  out: bf16 [Nd, Kd] (transposed)
__global__ void k_cvtT_bf16(const float* __restrict__ in, __bf16* __restrict__ out,
                            int Kd, int Nd) {
  int i = blockIdx.x * blockDim.x + threadIdx.x;
  if (i < Kd * Nd) {
    int k = i / Nd, n = i - k * Nd;
    out[(size_t)n * Kd + k] = (__bf16)in[i];
  }
}

// ---------------- register-blocked 32x64 wave-tile GEMM core ----------------
// A: [M,Kd] row-major bf16, Bt: [N,Kd] row-major bf16 (B transposed).
// Each wave produces a 32(M) x 64(N) tile: acc[i][j] is the (16i, 16j) subtile.
// Per 32-wide k-step: 4 A-chunk + 8 B-chunk b128 loads feed 8 WMMAs.
static __device__ __forceinline__ void gemm_block(const __bf16* __restrict__ A, int lda,
                                                  const __bf16* __restrict__ Bt, int ldb,
                                                  int m0, int n0, int Kd, v8f acc[2][4]) {
  const int lane = threadIdx.x & 31;
  const int l15  = lane & 15;
  const int koff = (lane >> 4) << 3;
  const __bf16* arow0 = A + (size_t)(m0 + l15) * lda + koff;
  const __bf16* arow1 = A + (size_t)(m0 + 16 + l15) * lda + koff;
  const __bf16* brow[4];
#pragma unroll
  for (int j = 0; j < 4; ++j)
    brow[j] = Bt + (size_t)(n0 + 16 * j + l15) * ldb + koff;

  for (int k = 0; k < Kd; k += 32) {
    Frag a0, a1, b[4];
    a0.h8[0] = *(const v8bf*)(arow0 + k);
    a0.h8[1] = *(const v8bf*)(arow0 + k + 16);
    a1.h8[0] = *(const v8bf*)(arow1 + k);
    a1.h8[1] = *(const v8bf*)(arow1 + k + 16);
#pragma unroll
    for (int j = 0; j < 4; ++j) {
      b[j].h8[0] = *(const v8bf*)(brow[j] + k);
      b[j].h8[1] = *(const v8bf*)(brow[j] + k + 16);
    }
    __builtin_prefetch(arow0 + k + 128, 0, 1);
    __builtin_prefetch(arow1 + k + 128, 0, 1);
#pragma unroll
    for (int j = 0; j < 4; ++j) {
      acc[0][j] = wmma_bf16(a0, b[j], acc[0][j]);
      acc[1][j] = wmma_bf16(a1, b[j], acc[1][j]);
    }
  }
}

static __device__ __forceinline__ void zero_acc(v8f acc[2][4]) {
  v8f z = {};
#pragma unroll
  for (int i = 0; i < 2; ++i)
#pragma unroll
    for (int j = 0; j < 4; ++j) acc[i][j] = z;
}

// ---------------- Q projection: Q = hs @ Wq -> [B,H,S,D] bf16 ----------------
__global__ __launch_bounds__(256) void k_gemm_q(const __bf16* __restrict__ A,
                                                const __bf16* __restrict__ Bt,
                                                __bf16* __restrict__ Qout) {
  const int lane = threadIdx.x & 31;
  const int tile = blockIdx.x * 8 + (threadIdx.x >> 5);
  const int tn = tile & 15;   // EE/64 = 16 N-tiles
  const int tm = tile >> 4;   // (B*S)/32 = 128 M-tiles
  v8f acc[2][4];
  zero_acc(acc);
  gemm_block(A, EE, Bt, EE, tm * 32, tn * 64, EE, acc);
  const int l15 = lane & 15;
  const int msub = (lane < 16) ? 0 : 8;
#pragma unroll
  for (int i = 0; i < 2; ++i)
#pragma unroll
    for (int j = 0; j < 4; ++j) {
      int n = tn * 64 + j * 16 + l15;
      int h = n >> 6, d = n & 63;
#pragma unroll
      for (int r = 0; r < 8; ++r) {
        int m = tm * 32 + i * 16 + msub + r;
        int b = m >> 11, s = m & (SS - 1);
        Qout[((size_t)(b * HH + h) * SS + s) * DD + d] = (__bf16)acc[i][j][r];
      }
    }
}

// ---------------- KV projection: kv = kvs @ Wkv; K->[B,H,K,D], V->[B,H,D,K] ----------------
__global__ __launch_bounds__(256) void k_gemm_kv(const __bf16* __restrict__ A,
                                                 const __bf16* __restrict__ Bt,
                                                 __bf16* __restrict__ Kout,
                                                 __bf16* __restrict__ VtOut) {
  const int lane = threadIdx.x & 31;
  const int tile = blockIdx.x * 8 + (threadIdx.x >> 5);
  const int tn = tile & 31;   // 2*EE/64 = 32 N-tiles
  const int tm = tile >> 5;   // (B*K)/32 = 128 M-tiles
  v8f acc[2][4];
  zero_acc(acc);
  gemm_block(A, EE, Bt, EE, tm * 32, tn * 64, EE, acc);
  const int l15 = lane & 15;
  const int msub = (lane < 16) ? 0 : 8;
#pragma unroll
  for (int i = 0; i < 2; ++i)
#pragma unroll
    for (int j = 0; j < 4; ++j) {
      int n = tn * 64 + j * 16 + l15;       // [0, 2E)
      int isV = n >> 10;
      int rem = n & 1023;
      int h = rem >> 6, d = rem & 63;
#pragma unroll
      for (int r = 0; r < 8; ++r) {
        int m = tm * 32 + i * 16 + msub + r;
        int b = m >> 11, kq = m & (KVN - 1);
        __bf16 val = (__bf16)acc[i][j][r];
        size_t bh = (size_t)(b * HH + h);
        if (isV) VtOut[(bh * DD + d) * KVN + kq] = val;
        else     Kout[(bh * KVN + kq) * DD + d] = val;
      }
    }
}

// ---------------- Output projection: out = Attn @ Wo -> fp32 ----------------
__global__ __launch_bounds__(256) void k_gemm_out(const __bf16* __restrict__ A,
                                                  const __bf16* __restrict__ Bt,
                                                  float* __restrict__ Out) {
  const int lane = threadIdx.x & 31;
  const int tile = blockIdx.x * 8 + (threadIdx.x >> 5);
  const int tn = tile & 15;
  const int tm = tile >> 4;
  v8f acc[2][4];
  zero_acc(acc);
  gemm_block(A, EE, Bt, EE, tm * 32, tn * 64, EE, acc);
  const int l15 = lane & 15;
  const int msub = (lane < 16) ? 0 : 8;
#pragma unroll
  for (int i = 0; i < 2; ++i)
#pragma unroll
    for (int j = 0; j < 4; ++j) {
      int n = tn * 64 + j * 16 + l15;
#pragma unroll
      for (int r = 0; r < 8; ++r) {
        int m = tm * 32 + i * 16 + msub + r;
        Out[(size_t)m * EE + n] = acc[i][j][r];
      }
    }
}

// ---------------- T5 relative-position bias (exact bucketing) ----------------
static __device__ __forceinline__ float t5_bias(int rel, int h, const float* __restrict__ rb) {
  const int nb = NBK / 2;        // 16
  const int max_exact = nb / 2;  // 8
  int bucket = (rel > 0) ? nb : 0;
  int ar = (rel < 0) ? -rel : rel;
  int add;
  if (ar < max_exact) {
    add = ar;
  } else {
    // max_exact + trunc( log(ar/8)/log(128/8) * (nb-max_exact) ); 8/ln(16)=2.88539004
    int large = max_exact + (int)(__logf((float)ar * 0.125f) * 2.88539004f);
    add = (large < nb - 1) ? large : (nb - 1);
  }
  return rb[(bucket + add) * HH + h];
}

// ---------------- fused flash attention (one 16-row s-tile per wave) ----------------
__global__ __launch_bounds__(128) void k_flash(const __bf16* __restrict__ Q,
                                               const __bf16* __restrict__ Km,
                                               const __bf16* __restrict__ Vt,
                                               const float* __restrict__ rb,
                                               __bf16* __restrict__ Aout) {
  __shared__ __align__(16) __bf16 lds[4][16][32];   // per-wave P tile
  const int wv   = threadIdx.x >> 5;
  const int lane = threadIdx.x & 31;
  const int l15  = lane & 15;
  const int koff = (lane >> 4) << 3;
  const int tile = blockIdx.x * 4 + wv;             // B*H*(S/16) = 4096 tiles
  const int st = tile & (SS / 16 - 1);
  const int bh = tile >> 7;
  const int h = bh & (HH - 1);
  const int b = bh >> 4;
  const int s0 = st * 16;

  // Q fragments for the full D=64 (two K=32 chunks)
  const __bf16* qrow = Q + ((size_t)bh * SS + s0 + l15) * DD + koff;
  Frag qf0, qf1;
  qf0.h8[0] = *(const v8bf*)(qrow);
  qf0.h8[1] = *(const v8bf*)(qrow + 16);
  qf1.h8[0] = *(const v8bf*)(qrow + 32);
  qf1.h8[1] = *(const v8bf*)(qrow + 48);

  const __bf16* kbase = Km + (size_t)bh * KVN * DD;
  const __bf16* vbase = Vt + (size_t)bh * DD * KVN;

  float mi[8], li[8];
  v8f vzero = {};
  v8f o[4];
#pragma unroll
  for (int j = 0; j < 4; ++j) o[j] = vzero;
#pragma unroll
  for (int r = 0; r < 8; ++r) { mi[r] = -3.0e38f; li[r] = 0.f; }

  const int mrow0 = s0 + ((lane < 16) ? 0 : 8);
  const int mloc  = (lane < 16) ? 0 : 8;

  for (int kv0 = 0; kv0 < KVN; kv0 += 32) {
    // ---- scores: two 16x16 tiles via QK^T ----
    v8f sc[2];
#pragma unroll
    for (int t = 0; t < 2; ++t) {
      const __bf16* krow = kbase + (size_t)(kv0 + t * 16 + l15) * DD + koff;
      Frag k0, k1;
      k0.h8[0] = *(const v8bf*)(krow);
      k0.h8[1] = *(const v8bf*)(krow + 16);
      k1.h8[0] = *(const v8bf*)(krow + 32);
      k1.h8[1] = *(const v8bf*)(krow + 48);
      v8f c = vzero;
      c = wmma_bf16(qf0, k0, c);
      c = wmma_bf16(qf1, k1, c);
      sc[t] = c;
    }
    // ---- add T5 relative bias ----
#pragma unroll
    for (int t = 0; t < 2; ++t) {
      int ncol = kv0 + t * 16 + l15;
#pragma unroll
      for (int r = 0; r < 8; ++r)
        sc[t][r] += t5_bias(ncol - (mrow0 + r), h, rb);
    }
    // ---- online softmax (row reductions across 16-lane halves) ----
    float mc[8];
#pragma unroll
    for (int r = 0; r < 8; ++r) mc[r] = fmaxf(sc[0][r], sc[1][r]);
#pragma unroll
    for (int off = 1; off < 16; off <<= 1)
#pragma unroll
      for (int r = 0; r < 8; ++r) mc[r] = fmaxf(mc[r], __shfl_xor(mc[r], off, 32));
    float corr[8];
#pragma unroll
    for (int r = 0; r < 8; ++r) {
      float mn = fmaxf(mi[r], mc[r]);
      corr[r] = __expf(mi[r] - mn);
      mi[r] = mn;
    }
    float rs[8];
#pragma unroll
    for (int r = 0; r < 8; ++r) rs[r] = 0.f;
#pragma unroll
    for (int t = 0; t < 2; ++t)
#pragma unroll
      for (int r = 0; r < 8; ++r) {
        float p = __expf(sc[t][r] - mi[r]);
        sc[t][r] = p;
        rs[r] += p;
      }
#pragma unroll
    for (int off = 1; off < 16; off <<= 1)
#pragma unroll
      for (int r = 0; r < 8; ++r) rs[r] += __shfl_xor(rs[r], off, 32);
#pragma unroll
    for (int r = 0; r < 8; ++r) li[r] = li[r] * corr[r] + rs[r];
#pragma unroll
    for (int j = 0; j < 4; ++j)
#pragma unroll
      for (int r = 0; r < 8; ++r) o[j][r] *= corr[r];

    // ---- P: C-layout -> LDS -> A-fragment layout (bf16) ----
#pragma unroll
    for (int t = 0; t < 2; ++t)
#pragma unroll
      for (int r = 0; r < 8; ++r)
        lds[wv][mloc + r][t * 16 + l15] = (__bf16)sc[t][r];
    Frag pf;
    pf.h8[0] = *(const v8bf*)&lds[wv][l15][koff];
    pf.h8[1] = *(const v8bf*)&lds[wv][l15][16 + koff];

    // ---- O += P @ V  (V stored transposed: contiguous fragment loads) ----
#pragma unroll
    for (int j = 0; j < 4; ++j) {
      const __bf16* vrow = vbase + (size_t)(j * 16 + l15) * KVN + kv0 + koff;
      Frag vf;
      vf.h8[0] = *(const v8bf*)(vrow);
      vf.h8[1] = *(const v8bf*)(vrow + 16);
      o[j] = wmma_bf16(pf, vf, o[j]);
    }
  }
  // ---- epilogue: normalize + store [B,S,H*D] bf16 ----
#pragma unroll
  for (int j = 0; j < 4; ++j)
#pragma unroll
    for (int r = 0; r < 8; ++r) {
      float val = o[j][r] / li[r];
      int srow = mrow0 + r;
      int col = h * DD + j * 16 + l15;
      Aout[((size_t)b * SS + srow) * EE + col] = (__bf16)val;
    }
}

// ---------------- launcher ----------------
extern "C" void kernel_launch(void* const* d_in, const int* in_sizes, int n_in,
                              void* d_out, int out_size, void* d_ws, size_t ws_size,
                              hipStream_t stream) {
  const float* hs  = (const float*)d_in[0];
  const float* kvs = (const float*)d_in[1];
  const float* Wq  = (const float*)d_in[2];
  const float* Wkv = (const float*)d_in[3];
  const float* Wo  = (const float*)d_in[4];
  const float* rb  = (const float*)d_in[5];
  float* out = (float*)d_out;

  char* p = (char*)d_ws;
  const size_t nTok = (size_t)BB * SS * EE;   // 4M elems
  __bf16* hs_bf  = (__bf16*)p; p += nTok * 2;
  __bf16* kvs_bf = (__bf16*)p; p += nTok * 2;
  __bf16* WqT    = (__bf16*)p; p += (size_t)EE * EE * 2;
  __bf16* WkvT   = (__bf16*)p; p += (size_t)EE * 2 * EE * 2;
  __bf16* WoT    = (__bf16*)p; p += (size_t)EE * EE * 2;
  __bf16* Qbf    = (__bf16*)p; p += nTok * 2;  // [B,H,S,D]
  __bf16* Kbf    = (__bf16*)p; p += nTok * 2;  // [B,H,K,D]
  __bf16* Vtbf   = (__bf16*)p; p += nTok * 2;  // [B,H,D,K]
  __bf16* Abf    = (__bf16*)p; p += nTok * 2;  // [B*S, E]

  k_cvt_bf16<<<(int)(nTok / 256), 256, 0, stream>>>(hs, hs_bf, (int)nTok);
  k_cvt_bf16<<<(int)(nTok / 256), 256, 0, stream>>>(kvs, kvs_bf, (int)nTok);
  k_cvtT_bf16<<<(EE * EE) / 256, 256, 0, stream>>>(Wq, WqT, EE, EE);
  k_cvtT_bf16<<<(EE * 2 * EE) / 256, 256, 0, stream>>>(Wkv, WkvT, EE, 2 * EE);
  k_cvtT_bf16<<<(EE * EE) / 256, 256, 0, stream>>>(Wo, WoT, EE, EE);

  // 32x64 tiles: q/out -> (4096/32)*(1024/64)=2048 wave-tiles / 8 waves = 256 blocks
  //              kv    -> (4096/32)*(2048/64)=4096 wave-tiles / 8 waves = 512 blocks
  k_gemm_q  <<<256, 256, 0, stream>>>(hs_bf, WqT, Qbf);
  k_gemm_kv <<<512, 256, 0, stream>>>(kvs_bf, WkvT, Kbf, Vtbf);
  k_flash   <<<1024, 128, 0, stream>>>(Qbf, Kbf, Vtbf, rb, Abf);
  k_gemm_out<<<256, 256, 0, stream>>>(Abf, WoT, out);
}
